// AttentionLayer_71794673320329
// MI455X (gfx1250) — matrix operand order repaired and checked by reading, not compiled, not run on hardware
//
#include <hip/hip_runtime.h>

#define NB 4
#define NS 2048
#define NH 16
#define NDH 64
#define ND 1024
#define SCP 2056   // padded LDS row stride (f16): 4112 B = 1028 dw = 4 banks shift/row

typedef _Float16 h16;
typedef __attribute__((ext_vector_type(16))) _Float16 v16h;
typedef __attribute__((ext_vector_type(8)))  _Float16 v8h;
typedef __attribute__((ext_vector_type(4)))  _Float16 v4h;
typedef __attribute__((ext_vector_type(8)))  float    v8f;

__device__ __forceinline__ v16h cat8(v8h lo, v8h hi) {
  return __builtin_shufflevector(lo, hi, 0,1,2,3,4,5,6,7,8,9,10,11,12,13,14,15);
}

// f32 -> f16 elementwise, 4 elements/thread, fully coalesced.
__global__ __launch_bounds__(256)
void xcvt_kernel(const float* __restrict__ x, h16* __restrict__ xh) {
  size_t idx = (size_t)blockIdx.x * 256 + threadIdx.x;
  float4 v4 = *(const float4*)(x + idx * 4);
  v4h r;
  r[0] = (h16)v4.x; r[1] = (h16)v4.y; r[2] = (h16)v4.z; r[3] = (h16)v4.w;
  *(v4h*)(xh + idx * 4) = r;
}

// Transpose + convert weights via LDS tile: wt[n][k] = (f16) w[k][n]
__global__ __launch_bounds__(256)
void wt_cvt_kernel(const float* __restrict__ w, h16* __restrict__ wt) {
  __shared__ float t[32][33];
  int tx = threadIdx.x & 31, ty = threadIdx.x >> 5;   // 32 x 8
  int k0 = blockIdx.x * 32, n0 = blockIdx.y * 32;
  #pragma unroll
  for (int i = 0; i < 4; i++) {
    int kk = ty + i * 8;
    t[kk][tx] = w[(size_t)(k0 + kk) * ND + n0 + tx];  // coalesced over n
  }
  __syncthreads();
  #pragma unroll
  for (int i = 0; i < 4; i++) {
    int nn = ty + i * 8;
    wt[(size_t)(n0 + nn) * ND + k0 + tx] = (h16)t[tx][nn];  // coalesced over k
  }
}

// GEMM: Out[m][n] = sum_k Xh[m][k]*W[k][n] + bias[n]
// Xh f16 row-major [B*S][D]; Wt f16 pre-transposed [n][k].
// Wave w owns 32M x 64N: 8 accumulators, 2 A-frags reused across 4 B-frags/k-step.
__global__ __launch_bounds__(128)
void proj_kernel(const h16* __restrict__ Xh, const h16* __restrict__ Wt,
                 const float* __restrict__ bias,
                 h16* __restrict__ OutRM, h16* __restrict__ OutT, int is_v) {
  const int lane = threadIdx.x & 31;
  const int w    = threadIdx.x >> 5;
  const int llo  = lane & 15, lhi = lane >> 4;
  const int m_base = blockIdx.x * 128 + w * 32;
  const int n_base = blockIdx.y * 64;

  const h16* xrow0 = Xh + (size_t)(m_base + llo) * ND + lhi * 16;      // A: M = llo
  const h16* xrow1 = Xh + (size_t)(m_base + 16 + llo) * ND + lhi * 16;
  const h16* wrow[4];
  #pragma unroll
  for (int j = 0; j < 4; j++)                                          // B: N = llo
    wrow[j] = Wt + (size_t)(n_base + j * 16 + llo) * ND + lhi * 16;

  v8f acc[8];
  #pragma unroll
  for (int j = 0; j < 8; j++) acc[j] = (v8f){0.f,0.f,0.f,0.f,0.f,0.f,0.f,0.f};

  for (int k0 = 0; k0 < ND; k0 += 32) {
    v16h a0 = cat8(*(const v8h*)(xrow0 + k0), *(const v8h*)(xrow0 + k0 + 8));
    v16h a1 = cat8(*(const v8h*)(xrow1 + k0), *(const v8h*)(xrow1 + k0 + 8));
    #pragma unroll
    for (int j = 0; j < 4; j++) {
      v16h b = *(const v16h*)(wrow[j] + k0);  // B: k = k0 + lhi*16 + e
      acc[j]     = __builtin_amdgcn_wmma_f32_16x16x32_f16(false, a0, false, b,
                                                          (short)0, acc[j],     false, false);
      acc[4 + j] = __builtin_amdgcn_wmma_f32_16x16x32_f16(false, a1, false, b,
                                                          (short)0, acc[4 + j], false, false);
    }
  }
  #pragma unroll
  for (int mt = 0; mt < 2; mt++) {
    #pragma unroll
    for (int j = 0; j < 4; j++) {
      int nb = n_base + j * 16;
      float bvj = bias[nb + llo];
      const v8f& a = acc[mt * 4 + j];
      if (!is_v) {
        #pragma unroll
        for (int v = 0; v < 8; v++) {               // C/D: row = lhi*8+v, col = llo
          int m = m_base + mt * 16 + lhi * 8 + v;
          OutRM[(size_t)m * ND + nb + llo] = (h16)(a[v] + bvj);
        }
      } else {
        int n = nb + llo;
        int h = n >> 6, dh = n & 63;
        #pragma unroll
        for (int v = 0; v < 8; v++) {
          int m = m_base + mt * 16 + lhi * 8 + v;
          int b_ = m >> 11, s = m & 2047;
          OutT[(((size_t)b_ * NH + h) * NDH + dh) * NS + s] = (h16)(a[v] + bvj);
        }
      }
    }
  }
}

// One WG = one (b, h, 16-row q block). Scores row-block lives in padded LDS.
__global__ __launch_bounds__(128)
void attn_kernel(const h16* __restrict__ Qp, const h16* __restrict__ Kp,
                 const h16* __restrict__ Vt, const int* __restrict__ mask,
                 float* __restrict__ out) {
  __shared__ __align__(16) h16 sc[16][SCP];         // 16 x 2056 f16 = 65792 B
  __shared__ float rowinv[16];
  const int lane = threadIdx.x & 31;
  const int w    = threadIdx.x >> 5;
  const int llo  = lane & 15, lhi = lane >> 4;
  const int qb = blockIdx.x, hh = blockIdx.y, bb = blockIdx.z;

  // ---- Phase 1: scores = (Q . K^T) * 1/sqrt(64) + mask-adder -> LDS(f16)
  const h16* qrow = Qp + ((size_t)(bb * NS + qb * 16 + llo)) * ND + hh * 64;
  v16h aq0 = cat8(*(const v8h*)(qrow +      lhi * 16), *(const v8h*)(qrow + 8 + lhi * 16));
  v16h aq1 = cat8(*(const v8h*)(qrow + 32 + lhi * 16), *(const v8h*)(qrow + 40 + lhi * 16));
  for (int kt = w; kt < NS / 16; kt += 4) {
    const h16* krow = Kp + ((size_t)(bb * NS + kt * 16 + llo)) * ND + hh * 64;
    int ktn = (kt + 4) & 127;                        // stream-prefetch next K tile
    __builtin_prefetch((const void*)(Kp + ((size_t)(bb * NS + ktn * 16 + llo)) * ND + hh * 64), 0, 1);
    v16h bk0 = *(const v16h*)(krow +      lhi * 16); // K row-major == B for Q.K^T
    v16h bk1 = *(const v16h*)(krow + 32 + lhi * 16);
    v8f z = {0.f,0.f,0.f,0.f,0.f,0.f,0.f,0.f};
    z = __builtin_amdgcn_wmma_f32_16x16x32_f16(false, aq0, false, bk0, (short)0, z, false, false);
    z = __builtin_amdgcn_wmma_f32_16x16x32_f16(false, aq1, false, bk1, (short)0, z, false, false);
    int sk = kt * 16 + llo;
    #pragma unroll
    for (int v = 0; v < 8; v++) {
      int row = lhi * 8 + v;
      int sq  = qb * 16 + row;
      int mv  = mask[((size_t)bb * NS + sq) * NS + sk];
      sc[row][sk] = (h16)(z[v] * 0.125f + (1.0f - (float)mv) * -10000.0f);
    }
  }
  __syncthreads();

  // ---- Phase 2: softmax (unnormalized): 8 lanes/row, v8h-vectorized passes
  {
    int row = threadIdx.x >> 3;
    int g   = threadIdx.x & 7;
    float m = -3.0e38f;
    for (int i = 0; i < NS / 64; i++) {
      v8h x8 = *(const v8h*)&sc[row][g * 8 + i * 64];
      #pragma unroll
      for (int e = 0; e < 8; e++) m = fmaxf(m, (float)x8[e]);
    }
    m = fmaxf(m, __shfl_xor(m, 1, 8));
    m = fmaxf(m, __shfl_xor(m, 2, 8));
    m = fmaxf(m, __shfl_xor(m, 4, 8));
    float s = 0.f;
    for (int i = 0; i < NS / 64; i++) {
      h16* p = &sc[row][g * 8 + i * 64];
      v8h x8 = *(const v8h*)p;
      v8h e8;
      #pragma unroll
      for (int e = 0; e < 8; e++) {
        float ev = __expf((float)x8[e] - m);
        e8[e] = (h16)ev;           // e <= 1, f16-safe; normalization deferred
        s += ev;
      }
      *(v8h*)p = e8;
    }
    s += __shfl_xor(s, 1, 8);
    s += __shfl_xor(s, 2, 8);
    s += __shfl_xor(s, 4, 8);
    if (g == 0) rowinv[row] = 1.0f / s;
  }
  __syncthreads();

  // ---- Phase 3: ctx = P (16x2048, LDS) x V_head (2048x64); wave w owns dh slice
  v8f acc = {0.f,0.f,0.f,0.f,0.f,0.f,0.f,0.f};
  const h16* vbase = Vt + (((size_t)bb * NH + hh) * NDH + w * 16 + llo) * NS;
  for (int ks = 0; ks < NS / 32; ks++) {
    v16h ap = cat8(*(const v8h*)&sc[llo][ks * 32 + lhi * 16],
                   *(const v8h*)&sc[llo][ks * 32 + 8 + lhi * 16]);
    v16h bv = *(const v16h*)(vbase + ks * 32 + lhi * 16); // Vt: K(=s) contiguous
    acc = __builtin_amdgcn_wmma_f32_16x16x32_f16(false, ap, false, bv,
                                                 (short)0, acc, false, false);
  }
  #pragma unroll
  for (int v = 0; v < 8; v++) {
    int r  = lhi * 8 + v;
    int sq = qb * 16 + r;
    out[((size_t)bb * NS + sq) * ND + hh * 64 + w * 16 + llo] = acc[v] * rowinv[r];
  }
}

extern "C" void kernel_launch(void* const* d_in, const int* in_sizes, int n_in,
                              void* d_out, int out_size, void* d_ws, size_t ws_size,
                              hipStream_t stream) {
  (void)in_sizes; (void)n_in; (void)out_size; (void)ws_size;
  const float* q  = (const float*)d_in[0];
  const float* k  = (const float*)d_in[1];
  const float* v  = (const float*)d_in[2];
  const int* mask = (const int*)d_in[3];
  const float* wq = (const float*)d_in[4];
  const float* bq = (const float*)d_in[5];
  const float* wk = (const float*)d_in[6];
  const float* bk = (const float*)d_in[7];
  const float* wv = (const float*)d_in[8];
  const float* bv = (const float*)d_in[9];
  float* out = (float*)d_out;

  char* ws = (char*)d_ws;
  const size_t MB = 1024ull * 1024ull;
  h16* wqt = (h16*)(ws + 0 * MB);        // 2 MB each
  h16* wkt = (h16*)(ws + 2 * MB);
  h16* wvt = (h16*)(ws + 4 * MB);
  h16* qh  = (h16*)(ws + 6 * MB);        // 16 MB each (B*S*D f16)
  h16* kh  = (h16*)(ws + 22 * MB);
  h16* vh  = (h16*)(ws + 38 * MB);
  h16* Qp  = (h16*)(ws + 54 * MB);
  h16* Kp  = (h16*)(ws + 70 * MB);
  h16* Vt  = (h16*)(ws + 86 * MB);       // [B][H][DH][S]

  const int NE = NB * NS * ND;           // 8388608 elements per tensor
  xcvt_kernel<<<NE / 1024, 256, 0, stream>>>(q, qh);
  xcvt_kernel<<<NE / 1024, 256, 0, stream>>>(k, kh);
  xcvt_kernel<<<NE / 1024, 256, 0, stream>>>(v, vh);

  dim3 wg(ND / 32, ND / 32);
  wt_cvt_kernel<<<wg, 256, 0, stream>>>(wq, wqt);
  wt_cvt_kernel<<<wg, 256, 0, stream>>>(wk, wkt);
  wt_cvt_kernel<<<wg, 256, 0, stream>>>(wv, wvt);

  dim3 pg(NB * NS / 128, ND / 64);
  proj_kernel<<<pg, 128, 0, stream>>>(qh, wqt, bq, Qp, Qp, 0);
  proj_kernel<<<pg, 128, 0, stream>>>(kh, wkt, bk, Kp, Kp, 0);
  proj_kernel<<<pg, 128, 0, stream>>>(vh, wvt, bv, Vt, Vt, 1);

  dim3 ag(NS / 16, NH, NB);
  attn_kernel<<<ag, 128, 0, stream>>>(Qp, Kp, Vt, mask, out);
}
